// Correspondence_16200616640534
// MI455X (gfx1250) — matrix operand order repaired and checked by reading, not compiled, not run on hardware
//
#include <hip/hip_runtime.h>
#include <hip/hip_bf16.h>

typedef __attribute__((ext_vector_type(16))) __bf16 v16bf;
typedef __attribute__((ext_vector_type(8)))  float  v8f;
typedef __attribute__((ext_vector_type(4)))  int    v4i;
typedef __attribute__((address_space(1))) v4i gl_v4i;   // global
typedef __attribute__((address_space(3))) v4i ld_v4i;   // LDS

union Frag { uint4 q[2]; v16bf v; };

#define CLIPS 4
#define SNIP  8
#define CH    256
#define HW    784
#define THW   6272          // SNIP*HW
#define MT    392           // THW/16 tiles
#define PPOS  25088         // CLIPS*THW == 32*HW
#define PT    1568          // PPOS/16 tiles
#define SEGS  4
#define SEGMT 98            // MT/SEGS

#if __has_builtin(__builtin_amdgcn_global_load_async_to_lds_b128)
#define HAVE_ASYNC_LDS 1
#else
#define HAVE_ASYNC_LDS 0
#endif

__device__ __forceinline__ void wait_async_le4() {
#if HAVE_ASYNC_LDS
  asm volatile("s_wait_asynccnt 0x4" ::: "memory");
#endif
}
__device__ __forceinline__ void wait_async_0() {
#if HAVE_ASYNC_LDS
  asm volatile("s_wait_asynccnt 0x0" ::: "memory");
#endif
}

__device__ __forceinline__ unsigned short f2bf(float f) {
  unsigned u = __builtin_bit_cast(unsigned, f);
  u += 0x7fffu + ((u >> 16) & 1u);            // round-to-nearest-even
  return (unsigned short)(u >> 16);
}

// A-matrix 16x32 bf16 per-lane K offset (ISA 7.12.2): half0 holds K 0-7,16-23; half1 K 8-15,24-31
__device__ __forceinline__ int a_koff(int e, int half) {
  return (e < 8 ? e : e + 8) + half * 8;
}

// sorted top-5 insert (descending values, strict > keeps earliest index on ties)
__device__ __forceinline__ void ins5(float v, int id, float tv[5], int ti[5]) {
  if (v <= tv[4]) return;
  tv[4] = v; ti[4] = id;
#pragma unroll
  for (int k = 4; k > 0; --k) {
    if (tv[k] > tv[k - 1]) {
      float fv = tv[k]; tv[k] = tv[k - 1]; tv[k - 1] = fv;
      int   fi = ti[k]; ti[k] = ti[k - 1]; ti[k - 1] = fi;
    }
  }
}

// ---------------- stage 1: transpose x -> xsT[b][t][c] (fp32) ----------------
__global__ void __launch_bounds__(256) k_transpose(const float* __restrict__ x,
                                                   float* __restrict__ xsT) {
  int i  = blockIdx.x * 256 + threadIdx.x;      // i = bt*256 + c
  int c  = i & 255;
  int bt = i >> 8;
  int b  = bt / THW, t = bt % THW;
  int s  = t / HW,  hw = t % HW;
  xsT[i] = x[((size_t)(b * SNIP + s) * CH + c) * HW + hw];
}

// ---------------- stage 2: rnrm[bt] = rsqrt(sum_c xsT^2) ----------------
__global__ void __launch_bounds__(256) k_rnrm(const float* __restrict__ xsT,
                                              float* __restrict__ rnrm) {
  __shared__ float sred[256];
  int bt = blockIdx.x, c = threadIdx.x;
  float v = xsT[(size_t)bt * CH + c];
  sred[c] = v * v;
  __syncthreads();
#pragma unroll
  for (int s = 128; s > 0; s >>= 1) {
    if (c < s) sred[c] += sred[c + s];
    __syncthreads();
  }
  if (c == 0) rnrm[bt] = rsqrtf(sred[0]);
}

// ---------------- stage 3: pack WMMA-native bf16 tiles ----------------
__global__ void __launch_bounds__(256) k_apack(const float* __restrict__ xsT,
                                               unsigned short* __restrict__ Ap) {
  int i   = blockIdx.x * 256 + threadIdx.x;     // CLIPS*MT*8*512 elements
  int e   = i & 15, lane = (i >> 4) & 31, kc = (i >> 9) & 7;
  int mtb = i >> 12;
  int mt  = mtb % MT, b = mtb / MT;
  int t   = mt * 16 + (lane & 15);
  int c   = kc * 32 + a_koff(e, lane >> 4);
  Ap[i] = f2bf(xsT[(size_t)(b * THW + t) * CH + c]);
}

__global__ void __launch_bounds__(256) k_bpack(const float* __restrict__ xsT,
                                               unsigned short* __restrict__ Bp) {
  int i   = blockIdx.x * 256 + threadIdx.x;
  int e   = i & 15, lane = (i >> 4) & 31, kc = (i >> 9) & 7;
  int ntb = i >> 12;
  int nt  = ntb % MT, b = ntb / MT;
  int n   = nt * 16 + (lane & 15);
  int k   = kc * 32 + a_koff(e, lane >> 4);     // B 32x16: half0 K0-7,16-23; half1 K8-15,24-31
  Bp[i] = f2bf(xsT[(size_t)(b * THW + n) * CH + k]);
}

// ---------------- stage 4: fused Gram GEMM + streaming top-5 ----------------
// 4 waves share A tiles via async double-buffered LDS staging; B panel lives in VGPRs.
__global__ void __launch_bounds__(128) k_gram_topk(
    const unsigned short* __restrict__ Apack,
    const unsigned short* __restrict__ Bpack,
    const float* __restrict__ rnrm,
    float* __restrict__ pv, int* __restrict__ pi) {
  __shared__ __align__(16) unsigned short atile[2][4096];   // 2 x 8KB A tiles

  const int tid  = threadIdx.x;
  const int lane = tid & 31;
  const int wave = tid >> 5;
  const int half = lane >> 4;
  const int b    = blockIdx.y;
  const int seg  = blockIdx.z;
  const int nt   = blockIdx.x * 4 + wave;
  const int n    = nt * 16 + (lane & 15);
  const int blkn = (nt * 16) / HW;              // frame block: uniform per wave (784%16==0)

  // B panel (256 x 16 bf16) held in 64 VGPRs for the whole M sweep
  Frag breg[8];
#pragma unroll
  for (int kc = 0; kc < 8; ++kc) {
    const uint4* p = (const uint4*)(Bpack + (((size_t)(b * MT + nt) * 8 + kc) << 9) + lane * 16);
    breg[kc].q[0] = p[0]; breg[kc].q[1] = p[1];
  }
  const float rn_n = rnrm[b * THW + n];

  float tv[5]; int ti[5];
#pragma unroll
  for (int k = 0; k < 5; ++k) { tv[k] = -2.0f; ti[k] = 0; }

  const int mt0 = seg * SEGMT;

  // async-stage one 8KB A tile: 128 threads x 64B (4 x b128 each)
  auto issue_tile = [&](int mt, int buf) {
    const unsigned short* g = Apack + (((size_t)(b * MT + mt) * 8) << 9) + tid * 32;
    unsigned short* l = &atile[buf][tid * 32];
#if HAVE_ASYNC_LDS
#pragma unroll
    for (int u = 0; u < 4; ++u)
      __builtin_amdgcn_global_load_async_to_lds_b128(
          (gl_v4i*)(g + u * 8), (ld_v4i*)(l + u * 8), 0, 0);
#else
#pragma unroll
    for (int u = 0; u < 4; ++u)
      ((uint4*)l)[u] = ((const uint4*)g)[u];
#endif
  };

  issue_tile(mt0, 0);
  for (int i = 0; i < SEGMT; ++i) {
    const int mt  = mt0 + i;
    const int cur = i & 1;
    if (i + 1 < SEGMT) { issue_tile(mt + 1, cur ^ 1); wait_async_le4(); }
    else               { wait_async_0(); }
    __syncthreads();                             // tile mt visible to all 4 waves

    const int mblk = (mt * 16) / HW;             // uniform per tile (784%16==0)
    if (mblk != blkn) {                          // masked tiles: skip WMMA + top-k
      // stage all 8 A fragments in distinct registers: one DS clause, one wait,
      // then 8 back-to-back WMMAs (C-accumulate chain, no intra-chain waits)
      Frag af[8];
      const unsigned short* lbase = &atile[cur][lane * 16];
#pragma unroll
      for (int kc = 0; kc < 8; ++kc) {
        const uint4* lp = (const uint4*)(lbase + (kc << 9));
        af[kc].q[0] = lp[0]; af[kc].q[1] = lp[1];
      }
      v8f acc = {};
#pragma unroll
      for (int kc = 0; kc < 8; ++kc) {
        acc = __builtin_amdgcn_wmma_f32_16x16x32_bf16(false, af[kc].v, false, breg[kc].v,
                                                      (short)0, acc, false, false);
      }
      const int mbase = mt * 16 + 8 * half;      // D rows for this lane half
      const float4* rp = (const float4*)(rnrm + b * THW + mbase);
      float4 r0 = rp[0], r1 = rp[1];
      float rm[8] = {r0.x, r0.y, r0.z, r0.w, r1.x, r1.y, r1.z, r1.w};
#pragma unroll
      for (int r = 0; r < 8; ++r) {
        float cv = acc[r] * rm[r] * rn_n;
        ins5(cv, mbase + r, tv, ti);
      }
    }
    __syncthreads();                             // readers done before buf reuse
  }

  // merge the two half-wave partial top-5s for each column
  float ov[5]; int oi[5];
#pragma unroll
  for (int k = 0; k < 5; ++k) {
    ov[k] = __shfl(tv[k], lane + 16, 32);
    oi[k] = __shfl(ti[k], lane + 16, 32);
  }
  if (lane < 16) {
#pragma unroll
    for (int k = 0; k < 5; ++k) ins5(ov[k], oi[k], tv, ti);
    size_t base = ((size_t)((seg * CLIPS + b) * THW) + n) * 5;
#pragma unroll
    for (int k = 0; k < 5; ++k) { pv[base + k] = tv[k]; pi[base + k] = ti[k]; }
  }
}

// ---------------- stage 5: merge the 4 M-segments ----------------
__global__ void __launch_bounds__(256) k_topk_merge(const float* __restrict__ pv,
                                                    const int* __restrict__ pi,
                                                    int* __restrict__ idxf) {
  int tid = blockIdx.x * 256 + threadIdx.x;
  if (tid >= CLIPS * THW) return;
  int b = tid / THW, j = tid % THW;
  float tv[5]; int ti[5];
#pragma unroll
  for (int k = 0; k < 5; ++k) { tv[k] = -2.0f; ti[k] = 0; }
  for (int seg = 0; seg < SEGS; ++seg) {
    size_t base = ((size_t)((seg * CLIPS + b) * THW) + j) * 5;
#pragma unroll
    for (int k = 0; k < 5; ++k) ins5(pv[base + k], pi[base + k], tv, ti);
  }
#pragma unroll
  for (int k = 0; k < 5; ++k) idxf[(size_t)tid * 5 + k] = ti[k];
}

// ---------------- stage 6: gather + max over top-5 (fp32) ----------------
__global__ void __launch_bounds__(256) k_gather_max(const float* __restrict__ xsT,
                                                    const int* __restrict__ idxf,
                                                    float* __restrict__ yT) {
  int bt = blockIdx.x;                  // b*THW + t
  int c  = threadIdx.x;
  int b  = bt / THW;
  const size_t rowb = (size_t)b * THW * CH;
  float v = -3.4e38f;
#pragma unroll
  for (int k = 0; k < 5; ++k) {
    int id = idxf[(size_t)bt * 5 + k];  // uniform per block -> scalar load
    v = fmaxf(v, xsT[rowb + (size_t)id * CH + c]);
  }
  yT[(size_t)bt * CH + c] = v;
}

// ---------------- stage 7: BN batch stats ----------------
__global__ void k_init(float* __restrict__ stats) {
  stats[threadIdx.x] = 0.0f;            // 512 threads: sum + sumsq
}

__global__ void __launch_bounds__(256) k_stats(const float* __restrict__ yT,
                                               float* __restrict__ stats) {
  int c = threadIdx.x;
  float s = 0.0f, q = 0.0f;
  for (int g = blockIdx.x; g < PPOS; g += gridDim.x) {
    float v = yT[(size_t)g * CH + c];
    s += v; q += v * v;
  }
  atomicAdd(&stats[c], s);
  atomicAdd(&stats[CH + c], q);
}

__global__ void k_bnfin(const float* __restrict__ stats,
                        const float* __restrict__ gamma,
                        const float* __restrict__ beta,
                        float* __restrict__ scsh) {
  int c = threadIdx.x;
  float mean = stats[c] * (1.0f / PPOS);
  float var  = stats[CH + c] * (1.0f / PPOS) - mean * mean;
  float sc   = gamma[c] * rsqrtf(var + 1e-5f);
  scsh[c]      = sc;
  scsh[CH + c] = beta[c] - mean * sc;
}

// ---------------- stage 8: BN+ReLU activation pack (bf16 A tiles) ----------------
__global__ void __launch_bounds__(256) k_actpack(const float* __restrict__ yT,
                                                 const float* __restrict__ scsh,
                                                 unsigned short* __restrict__ Ap2) {
  int i  = blockIdx.x * 256 + threadIdx.x;      // PT*8*512 elements
  int e  = i & 15, lane = (i >> 4) & 31, kc = (i >> 9) & 7;
  int mt = i >> 12;
  int p  = mt * 16 + (lane & 15);
  int c  = kc * 32 + a_koff(e, lane >> 4);
  float v = yT[(size_t)p * CH + c];
  v = fmaxf(scsh[c] * v + scsh[CH + c], 0.0f);
  Ap2[i] = f2bf(v);
}

__global__ void __launch_bounds__(256) k_wpack(const float* __restrict__ conv_w,
                                               unsigned short* __restrict__ Wp) {
  int i  = blockIdx.x * 256 + threadIdx.x;      // 16*8*512 elements
  int e  = i & 15, lane = (i >> 4) & 31, kc = (i >> 9) & 7;
  int nt = i >> 12;
  int o  = nt * 16 + (lane & 15);
  int k  = kc * 32 + a_koff(e, lane >> 4);
  Wp[i] = f2bf(conv_w[o * CH + k]);             // B[k][n] = W[o][k]
}

// ---------------- stage 9: 1x1 conv WMMA GEMM + bias + identity ----------------
__global__ void __launch_bounds__(128) k_conv_gemm(
    const unsigned short* __restrict__ Ap2,
    const unsigned short* __restrict__ Wp,
    const float* __restrict__ conv_b,
    const float* __restrict__ x_id,
    float* __restrict__ out) {
  const int lane = threadIdx.x & 31;
  const int wave = threadIdx.x >> 5;
  const int half = lane >> 4;
  const int nt   = blockIdx.x * 4 + wave;       // 0..15 output-channel tiles
  const int o    = nt * 16 + (lane & 15);

  Frag breg[8];
#pragma unroll
  for (int kc = 0; kc < 8; ++kc) {
    const uint4* p = (const uint4*)(Wp + (((nt * 8 + kc) << 9) + lane * 16));
    breg[kc].q[0] = p[0]; breg[kc].q[1] = p[1];
  }
  const float cb = conv_b[o];

  const int mt0 = blockIdx.y * 49;              // 32 groups of 49 tiles = 1568
  for (int mt = mt0; mt < mt0 + 49; ++mt) {
    v8f acc = {};
    const unsigned short* abase = Ap2 + (((size_t)mt * 8) << 9) + lane * 16;
#pragma unroll
    for (int kc = 0; kc < 8; ++kc) {
      Frag a;
      const uint4* ap = (const uint4*)(abase + (kc << 9));
      a.q[0] = ap[0]; a.q[1] = ap[1];
      acc = __builtin_amdgcn_wmma_f32_16x16x32_bf16(false, a.v, false, breg[kc].v,
                                                    (short)0, acc, false, false);
    }
    __builtin_prefetch(abase + (8 << 9), 0, 1);
    const int bs  = (mt * 16) / HW;             // uniform per tile (784%16==0)
    const int hw0 = (mt * 16) % HW + 8 * half;
    const size_t obase = ((size_t)bs * CH + o) * HW + hw0;
#pragma unroll
    for (int r = 0; r < 8; ++r) {
      out[obase + r] = acc[r] + cb + x_id[obase + r];
    }
  }
}

extern "C" void kernel_launch(void* const* d_in, const int* in_sizes, int n_in,
                              void* d_out, int out_size, void* d_ws, size_t ws_size,
                              hipStream_t stream) {
  const float* x      = (const float*)d_in[0];
  const float* gamma  = (const float*)d_in[1];
  const float* beta   = (const float*)d_in[2];
  const float* conv_w = (const float*)d_in[3];
  const float* conv_b = (const float*)d_in[4];
  (void)in_sizes; (void)n_in; (void)out_size; (void)ws_size;
  float* out = (float*)d_out;

  // workspace carve (256B aligned)
  char* w = (char*)d_ws;
  size_t off = 0;
  auto carve = [&](size_t bytes) -> void* {
    off = (off + 255) & ~(size_t)255;
    void* p = w + off;
    off += bytes;
    return p;
  };
  float*          xsT   = (float*)carve((size_t)PPOS * CH * 4);
  float*          rnrm  = (float*)carve((size_t)PPOS * 4);
  unsigned short* Apack = (unsigned short*)carve((size_t)CLIPS * MT * 8 * 512 * 2);
  unsigned short* Bpack = (unsigned short*)carve((size_t)CLIPS * MT * 8 * 512 * 2);
  float*          pv    = (float*)carve((size_t)SEGS * PPOS * 5 * 4);
  int*            pi    = (int*)carve((size_t)SEGS * PPOS * 5 * 4);
  int*            idxf  = (int*)carve((size_t)PPOS * 5 * 4);
  float*          yT    = (float*)carve((size_t)PPOS * CH * 4);
  float*          stats = (float*)carve(2 * CH * 4);
  float*          scsh  = (float*)carve(2 * CH * 4);
  unsigned short* Ap2   = (unsigned short*)carve((size_t)PT * 8 * 512 * 2);
  unsigned short* Wp    = (unsigned short*)carve((size_t)16 * 8 * 512 * 2);

  k_transpose<<<PPOS, 256, 0, stream>>>(x, xsT);
  k_rnrm<<<PPOS, 256, 0, stream>>>(xsT, rnrm);
  k_apack<<<PPOS, 256, 0, stream>>>(xsT, Apack);     // CLIPS*MT*8*512/256 = 25088
  k_bpack<<<PPOS, 256, 0, stream>>>(xsT, Bpack);
  k_gram_topk<<<dim3(98, CLIPS, SEGS), 128, 0, stream>>>(Apack, Bpack, rnrm, pv, pi);
  k_topk_merge<<<(PPOS + 255) / 256, 256, 0, stream>>>(pv, pi, idxf);
  k_gather_max<<<PPOS, 256, 0, stream>>>(xsT, idxf, yT);
  k_init<<<1, 512, 0, stream>>>(stats);
  k_stats<<<128, 256, 0, stream>>>(yT, stats);
  k_bnfin<<<1, 256, 0, stream>>>(stats, gamma, beta, scsh);
  k_actpack<<<PPOS, 256, 0, stream>>>(yT, scsh, Ap2); // PT*8*512/256 = 25088
  k_wpack<<<256, 256, 0, stream>>>(conv_w, Wp);
  k_conv_gemm<<<dim3(4, 32), 128, 0, stream>>>(Ap2, Wp, conv_b, x, out);
}
